// FusionMamba_25099788878250
// MI455X (gfx1250) — matrix operand order repaired and checked by reading, not compiled, not run on hardware
//
#include <hip/hip_runtime.h>
#include <hip/hip_bf16.h>

typedef __attribute__((ext_vector_type(16))) _Float16 v16h;
typedef __attribute__((ext_vector_type(8)))  float    v8f;

#define BN 4
#define LL 4096
#define DM 128

__device__ __forceinline__ float silu_f(float x) { return x / (1.0f + __expf(-x)); }
__device__ __forceinline__ float softplus_f(float x) {
  return (x > 30.0f) ? x : log1pf(__expf(x));
}

// ---------------------------------------------------------------------------
// WMMA GEMM:  C[M,N] = A[M,K] * B   (f16 inputs, f32 accumulate)
//   BNXK=true  : B given as W[N,K] row-major (C = A @ W^T)
//   BNXK=false : B given as B[K,N] row-major (C = A @ B)
// One wave per 16x32 output tile (two accumulators share one A fragment).
// K = KCNT*32, fully unrolled -> 2*KCNT back-to-back v_wmma per wave.
// M multiple of 16, N multiple of 32.
// ---------------------------------------------------------------------------
template <int KCNT, bool BNXK>
__global__ void gemm_wmma_kernel(const float* __restrict__ A, int lda,
                                 const float* __restrict__ Bm, int ldb,
                                 float* __restrict__ C, int ldc,
                                 int M, int N) {
  int gwave = (int)(((size_t)blockIdx.x * blockDim.x + threadIdx.x) >> 5);
  int lane  = threadIdx.x & 31;
  int tilesN = N >> 5;
  int tiles  = (M >> 4) * tilesN;
  if (gwave >= tiles) return;               // wave-uniform: EXEC stays all-ones
  int tm = gwave / tilesN;
  int tn = gwave - tm * tilesN;
  int m0 = tm << 4, n0 = tn << 5;
  int r  = lane & 15;
  int hi = lane >> 4;

  v8f acc0 = {}, acc1 = {};
  const float* Ar = A + (size_t)(m0 + r) * lda;
#pragma unroll
  for (int kt = 0; kt < KCNT; ++kt) {
    const int k0 = kt * 32;
    v16h a, b0, b1;
#pragma unroll
    for (int j = 0; j < 8; ++j) {
      a[j]     = (_Float16)Ar[k0 + hi * 16 + j];       // K = hi*16 + j
      a[j + 8] = (_Float16)Ar[k0 + hi * 16 + 8 + j];   // K = hi*16 + 8 + j
    }
    if (BNXK) {
      const float* Bp0 = Bm + (size_t)(n0 + r) * ldb + k0 + hi * 16;
      const float* Bp1 = Bp0 + (size_t)16 * ldb;
#pragma unroll
      for (int j = 0; j < 16; ++j) {                   // K = hi*16 + j
        b0[j] = (_Float16)Bp0[j];
        b1[j] = (_Float16)Bp1[j];
      }
    } else {
      const float* Bp = Bm + (size_t)(k0 + hi * 16) * ldb + (n0 + r);
#pragma unroll
      for (int j = 0; j < 16; ++j) {
        b0[j] = (_Float16)Bp[(size_t)j * ldb];
        b1[j] = (_Float16)Bp[(size_t)j * ldb + 16];
      }
    }
    acc0 = __builtin_amdgcn_wmma_f32_16x16x32_f16(false, a, false, b0,
                                                  (short)0, acc0, false, false);
    acc1 = __builtin_amdgcn_wmma_f32_16x16x32_f16(false, a, false, b1,
                                                  (short)0, acc1, false, false);
  }
#pragma unroll
  for (int v = 0; v < 8; ++v) {
    float* Cr = C + (size_t)(m0 + v + 8 * hi) * ldc + n0 + r;
    Cr[0]  = acc0[v];
    Cr[16] = acc1[v];
  }
}

// ---------------------------------------------------------------------------
// proj (depthwise 3x3 + pointwise 8->128 + silu) fused with 2x2 gated pooling.
// One block per output pixel (b, hh, ww); 128 threads = 128 output channels.
// ---------------------------------------------------------------------------
__global__ void proj_pool_kernel(const float* __restrict__ img,  // (B,8,128,128)
                                 const float* __restrict__ dw,   // (8,3,3)
                                 const float* __restrict__ pw,   // (128,8)
                                 const float* __restrict__ gate, // scalar
                                 float* __restrict__ out) {      // (B,4096,128)
  __shared__ float t[4][8];
  int blk = blockIdx.x;
  int b  = blk >> 12;
  int hh = (blk >> 6) & 63;
  int ww = blk & 63;
  int tid = threadIdx.x;
  if (tid < 32) {
    int pos = tid >> 3, c = tid & 7;
    int h = hh * 2 + (pos >> 1), w = ww * 2 + (pos & 1);
    const float* ip = img + ((size_t)b * 8 + c) * 128 * 128;
    float s = 0.0f;
    for (int dy = -1; dy <= 1; ++dy)
      for (int dx = -1; dx <= 1; ++dx) {
        int y = h + dy, x = w + dx;
        if (y >= 0 && y < 128 && x >= 0 && x < 128)
          s += ip[y * 128 + x] * dw[c * 9 + (dy + 1) * 3 + (dx + 1)];
      }
    t[pos][c] = s;
  }
  __syncthreads();
  float g = *gate;
  float vmax = -3.4e38f, vsum = 0.0f;
#pragma unroll
  for (int pos = 0; pos < 4; ++pos) {
    float acc = 0.0f;
#pragma unroll
    for (int c = 0; c < 8; ++c) acc += t[pos][c] * pw[tid * 8 + c];
    acc = silu_f(acc);
    vsum += acc;
    vmax = fmaxf(vmax, acc);
  }
  float v = g * (vsum * 0.25f) + (1.0f - g) * vmax;
  out[((size_t)b * 4096 + hh * 64 + ww) * DM + tid] = v;
}

// rmsnorm over 128-dim rows: y = x / (||x|| / sqrt(128) + eps) * w
__global__ void rmsnorm_kernel(const float* __restrict__ x, int ldx,
                               const float* __restrict__ w,
                               float* __restrict__ y, int ldy, int rows) {
  int wv   = (int)(((size_t)blockIdx.x * blockDim.x + threadIdx.x) >> 5);
  int lane = threadIdx.x & 31;
  if (wv >= rows) return;
  const float* xr = x + (size_t)wv * ldx;
  float v[4];
  float ss = 0.0f;
#pragma unroll
  for (int i = 0; i < 4; ++i) { v[i] = xr[lane + 32 * i]; ss += v[i] * v[i]; }
#pragma unroll
  for (int off = 16; off > 0; off >>= 1) ss += __shfl_xor(ss, off, 32);
  float nrm   = sqrtf(ss);
  float scale = 1.0f / (nrm * 0.08838834764831845f + 1e-6f); // 1/sqrt(128)
  float* yr = y + (size_t)wv * ldy;
#pragma unroll
  for (int i = 0; i < 4; ++i) yr[lane + 32 * i] = v[i] * scale * w[lane + 32 * i];
}

// depthwise 1D conv along L (+bias, +silu). flip=1: read x time-reversed per batch.
__global__ void dwconv_silu_kernel(const float* __restrict__ x, int ldx, int dim,
                                   const float* __restrict__ w, int k, int padL,
                                   const float* __restrict__ bias,
                                   float* __restrict__ y, int ldy, int flip) {
  size_t idx = (size_t)blockIdx.x * blockDim.x + threadIdx.x;
  size_t total = (size_t)BN * LL * dim;
  if (idx >= total) return;
  int d = (int)(idx % dim);
  int l = (int)((idx / dim) % LL);
  int b = (int)(idx / ((size_t)dim * LL));
  float s = bias[d];
  for (int j = 0; j < k; ++j) {
    int li = l + j - padL;
    if (li < 0 || li >= LL) continue;
    int lp = flip ? (LL - 1 - li) : li;
    s += x[((size_t)b * LL + lp) * ldx + d] * w[d * k + j];
  }
  y[((size_t)b * LL + l) * ldy + d] = silu_f(s);
}

// dbl[r,j] = xc[r,:] . x_proj[j,:]   (j < 24, K = 128)
__global__ void xproj_kernel(const float* __restrict__ xc,
                             const float* __restrict__ Wp,
                             float* __restrict__ dbl, size_t rows) {
  size_t idx = (size_t)blockIdx.x * blockDim.x + threadIdx.x;
  if (idx >= rows * 24) return;
  size_t r = idx / 24;
  int j = (int)(idx % 24);
  const float* xr = xc + r * DM;
  const float* wr = Wp + (size_t)j * DM;
  float s = 0.0f;
  for (int d = 0; d < DM; ++d) s += xr[d] * wr[d];
  dbl[r * 24 + j] = s;
}

// dt[r,d] = softplus(dbl[r,0:8] . dt_w[d,:] + dt_b[d])
__global__ void dt_kernel(const float* __restrict__ dbl,
                          const float* __restrict__ dtw,
                          const float* __restrict__ dtb,
                          float* __restrict__ dt, size_t rows) {
  size_t idx = (size_t)blockIdx.x * blockDim.x + threadIdx.x;
  if (idx >= rows * DM) return;
  size_t r = idx >> 7;
  int d = (int)(idx & 127);
  const float* br = dbl + r * 24;
  float s = dtb[d];
#pragma unroll
  for (int j = 0; j < 8; ++j) s += br[j] * dtw[d * 8 + j];
  dt[idx] = softplus_f(s);
}

// Selective scan: one thread per (b,d), n=8 state in registers, serial over L.
// flip=1: write y back in time-reversed (physical) order.
__global__ void scan_kernel(const float* __restrict__ xc,
                            const float* __restrict__ dt,
                            const float* __restrict__ dbl,
                            const float* __restrict__ alog,
                            const float* __restrict__ Dv,
                            float* __restrict__ ys, int flip) {
  int idx = blockIdx.x * blockDim.x + threadIdx.x;
  if (idx >= BN * DM) return;
  int b = idx >> 7, d = idx & 127;
  float a[8], h[8];
#pragma unroll
  for (int n = 0; n < 8; ++n) { a[n] = -__expf(alog[d * 8 + n]); h[n] = 0.0f; }
  float Dd = Dv[d];
  for (int l = 0; l < LL; ++l) {
    size_t r = (size_t)b * LL + l;
    float dtv = dt[r * DM + d];
    float u   = xc[r * DM + d];
    const float* bc = dbl + r * 24;
    float du = dtv * u;
    float yv = 0.0f;
#pragma unroll
    for (int n = 0; n < 8; ++n) {
      h[n] = __expf(dtv * a[n]) * h[n] + du * bc[8 + n];
      yv  += h[n] * bc[16 + n];
    }
    yv += u * Dd;
    int lp = flip ? (LL - 1 - l) : l;
    ys[((size_t)b * LL + lp) * DM + d] = yv;
  }
}

// y = 0.5*(yf+yb)*silu(z)
__global__ void combine_kernel(const float* __restrict__ yf,
                               const float* __restrict__ yb,
                               const float* __restrict__ z, int ldz,
                               float* __restrict__ y, size_t rows) {
  size_t idx = (size_t)blockIdx.x * blockDim.x + threadIdx.x;
  if (idx >= rows * DM) return;
  size_t r = idx >> 7;
  int d = (int)(idx & 127);
  float zv = silu_f(z[r * ldz + d]);
  y[idx] = 0.5f * (yf[idx] + yb[idx]) * zv;
}

// out = base + val * gamma   (gamma==nullptr -> 1)
__global__ void addscale_kernel(const float* __restrict__ base,
                                const float* __restrict__ val,
                                const float* __restrict__ gamma,
                                float* __restrict__ out, size_t rows) {
  size_t idx = (size_t)blockIdx.x * blockDim.x + threadIdx.x;
  if (idx >= rows * DM) return;
  int d = (int)(idx & 127);
  float g = gamma ? gamma[d] : 1.0f;
  out[idx] = base[idx] + val[idx] * g;
}

// Bilinear 64->128 upsample of both streams, concatenated to fused (B,256,128,128)
__global__ void fuse_resize_kernel(const float* __restrict__ spa,
                                   const float* __restrict__ spe,
                                   float* __restrict__ fused) {
  size_t idx = (size_t)blockIdx.x * blockDim.x + threadIdx.x;
  if (idx >= (size_t)BN * 256 * 128 * 128) return;
  int w = (int)(idx & 127);
  int h = (int)((idx >> 7) & 127);
  int c = (int)((idx >> 14) & 255);
  int b = (int)(idx >> 22);
  const float* src = (c < 128) ? spa : spe;
  int cc = c & 127;
  float sh = (h + 0.5f) * 0.5f - 0.5f;
  float sw = (w + 0.5f) * 0.5f - 0.5f;
  int h0 = (int)floorf(sh), w0 = (int)floorf(sw);
  float fh = sh - h0, fw = sw - w0;
  int h0c = min(max(h0, 0), 63), h1c = min(max(h0 + 1, 0), 63);
  int w0c = min(max(w0, 0), 63), w1c = min(max(w0 + 1, 0), 63);
  auto at = [&](int hy, int wx) {
    return src[((size_t)b * 4096 + hy * 64 + wx) * DM + cc];
  };
  float v = (1.0f - fh) * ((1.0f - fw) * at(h0c, w0c) + fw * at(h0c, w1c)) +
            fh * ((1.0f - fw) * at(h1c, w0c) + fw * at(h1c, w1c));
  fused[idx] = v;
}

// per-(b,c) spatial mean over 128x128
__global__ void chanmean_kernel(const float* __restrict__ fused,
                                float* __restrict__ mean) {
  __shared__ float sm[256];
  int bc = blockIdx.x;
  const float* p = fused + (size_t)bc * 16384;
  float s = 0.0f;
  for (int i = threadIdx.x; i < 16384; i += 256) s += p[i];
  sm[threadIdx.x] = s;
  __syncthreads();
  for (int off = 128; off > 0; off >>= 1) {
    if (threadIdx.x < off) sm[threadIdx.x] += sm[threadIdx.x + off];
    __syncthreads();
  }
  if (threadIdx.x == 0) mean[bc] = sm[0] * (1.0f / 16384.0f);
}

// ECA: attn[b,c] = sigmoid(w0*y[c-1] + w1*y[c] + w2*y[c+1])  (zero pad)
__global__ void eca_kernel(const float* __restrict__ mean,
                           const float* __restrict__ w,
                           float* __restrict__ attn) {
  int idx = blockIdx.x * blockDim.x + threadIdx.x;
  if (idx >= BN * 256) return;
  int b = idx >> 8, c = idx & 255;
  float ym = (c > 0) ? mean[b * 256 + c - 1] : 0.0f;
  float y0 = mean[idx];
  float yp = (c < 255) ? mean[b * 256 + c + 1] : 0.0f;
  float v = w[0] * ym + w[1] * y0 + w[2] * yp;
  attn[idx] = 1.0f / (1.0f + __expf(-v));
}

__global__ void scale_fused_kernel(float* __restrict__ fused,
                                   const float* __restrict__ attn) {
  size_t idx = (size_t)blockIdx.x * blockDim.x + threadIdx.x;
  if (idx >= (size_t)BN * 256 * 16384) return;
  int c = (int)((idx >> 14) & 255);
  int b = (int)(idx >> 22);
  fused[idx] *= attn[b * 256 + c];
}

// ---------------------------------------------------------------------------
// Host-side orchestration
// ---------------------------------------------------------------------------
struct MambaDirP { const float *cw, *cb, *xp, *dtw, *dtb, *alog, *dv; };
struct MambaP {
  const float* in_proj; const float* z_proj;
  MambaDirP f, b;
  const float* norm_w; const float* out_proj;
};

static MambaDirP dirp(void* const* d, int base) {
  return { (const float*)d[base],     (const float*)d[base + 1],
           (const float*)d[base + 2], (const float*)d[base + 3],
           (const float*)d[base + 4], (const float*)d[base + 5],
           (const float*)d[base + 6] };
}
static MambaP mambap(void* const* d, int base, bool cross) {
  MambaP p;
  p.in_proj = (const float*)d[base];
  int o = base + 1;
  if (cross) { p.z_proj = (const float*)d[o]; ++o; } else p.z_proj = nullptr;
  p.f = dirp(d, o);
  p.b = dirp(d, o + 7);
  p.norm_w   = (const float*)d[o + 14];
  p.out_proj = (const float*)d[o + 15];
  return p;
}

static inline int grid_e(size_t n) { return (int)((n + 255) / 256); }

static void launch_gemm(hipStream_t s, const float* A, int lda,
                        const float* Bm, int ldb, int bNxK,
                        float* C, int ldc, int M, int N, int K) {
  int tiles = (M >> 4) * (N >> 5);
  int blocks = (tiles + 7) / 8;  // 8 waves per 256-thread block
#define GEMM_DISPATCH(KC)                                                      \
  do {                                                                         \
    if (bNxK)                                                                  \
      gemm_wmma_kernel<KC, true><<<blocks, 256, 0, s>>>(A, lda, Bm, ldb, C,    \
                                                        ldc, M, N);            \
    else                                                                       \
      gemm_wmma_kernel<KC, false><<<blocks, 256, 0, s>>>(A, lda, Bm, ldb, C,   \
                                                         ldc, M, N);           \
  } while (0)
  switch (K >> 5) {
    case 2: GEMM_DISPATCH(2); break;   // K = 64
    case 4: GEMM_DISPATCH(4); break;   // K = 128
    case 8: GEMM_DISPATCH(8); break;   // K = 256
    default: break;                    // unused
  }
#undef GEMM_DISPATCH
}

static void run_mamba_core(hipStream_t s, const MambaP& p,
                           const float* xin, int ldxin,
                           const float* z, int ldz,
                           float* XC, float* DT, float* DBL,
                           float* YF, float* YB, float* Y, float* TMP) {
  const size_t rows = (size_t)BN * LL;
  for (int dir = 0; dir < 2; ++dir) {
    const MambaDirP& dp = dir ? p.b : p.f;
    float* ybuf = dir ? YB : YF;
    dwconv_silu_kernel<<<grid_e(rows * DM), 256, 0, s>>>(
        xin, ldxin, DM, dp.cw, 4, 3, dp.cb, XC, DM, dir);
    xproj_kernel<<<grid_e(rows * 24), 256, 0, s>>>(XC, dp.xp, DBL, rows);
    dt_kernel<<<grid_e(rows * DM), 256, 0, s>>>(DBL, dp.dtw, dp.dtb, DT, rows);
    scan_kernel<<<(BN * DM + 127) / 128, 128, 0, s>>>(XC, DT, DBL, dp.alog,
                                                      dp.dv, ybuf, dir);
  }
  combine_kernel<<<grid_e(rows * DM), 256, 0, s>>>(YF, YB, z, ldz, Y, rows);
  rmsnorm_kernel<<<(int)((rows * 32 + 255) / 256), 256, 0, s>>>(
      Y, DM, p.norm_w, TMP, DM, (int)rows);
  launch_gemm(s, TMP, DM, p.out_proj, DM, 1, Y, DM, (int)rows, DM, DM);
}

extern "C" void kernel_launch(void* const* d_in, const int* in_sizes, int n_in,
                              void* d_out, int out_size, void* d_ws, size_t ws_size,
                              hipStream_t stream) {
  (void)in_sizes; (void)n_in; (void)out_size; (void)ws_size;
  const size_t rows = (size_t)BN * LL;
  const size_t S1 = rows * DM;  // 2,097,152 floats

  // --- input leaves (setup_inputs() insertion order) ---
  const float* pan_in      = (const float*)d_in[0];
  const float* ms_in       = (const float*)d_in[1];
  const float* proj_pan_dw = (const float*)d_in[2];
  const float* proj_pan_pw = (const float*)d_in[3];
  const float* proj_ms_dw  = (const float*)d_in[4];
  const float* proj_ms_pw  = (const float*)d_in[5];
  const float* gate        = (const float*)d_in[6];
  MambaP p_shared = mambap(d_in, 7, false);          // 7..23
  const float* spa_norm  = (const float*)d_in[24];
  const float* spa_gamma = (const float*)d_in[25];
  const float* spe_norm  = (const float*)d_in[26];
  const float* spe_gamma = (const float*)d_in[27];
  // cross_spa: norm0=28, norm1=29, mamba=30..47, fc1=48, dw_w=49, dw_b=50, fc2=51
  const float* spaN0 = (const float*)d_in[28];
  const float* spaN1 = (const float*)d_in[29];
  MambaP p_cspa = mambap(d_in, 30, true);
  const float* spa_fc1 = (const float*)d_in[48];
  const float* spa_dww = (const float*)d_in[49];
  const float* spa_dwb = (const float*)d_in[50];
  const float* spa_fc2 = (const float*)d_in[51];
  // cross_spe: 52..75
  const float* speN0 = (const float*)d_in[52];
  const float* speN1 = (const float*)d_in[53];
  MambaP p_cspe = mambap(d_in, 54, true);
  const float* spe_fc1 = (const float*)d_in[72];
  const float* spe_dww = (const float*)d_in[73];
  const float* spe_dwb = (const float*)d_in[74];
  const float* spe_fc2 = (const float*)d_in[75];
  const float* eca_w    = (const float*)d_in[76];
  const float* out_conv = (const float*)d_in[77];

  // --- workspace layout (floats) ---
  float* W   = (float*)d_ws;
  float* PAN = W;                 // S1
  float* MS  = W + S1;            // S1
  float* SPA = W + 2 * S1;        // S1
  float* SPE = W + 3 * S1;        // S1
  float* XN  = W + 4 * S1;        // S1 (also holds BL x 64 ffn hidden)
  float* XZ  = W + 5 * S1;        // 2*S1  (x_in | z, ld 256)
  float* XC  = W + 7 * S1;        // S1
  float* DT  = W + 8 * S1;        // S1
  float* YF  = W + 9 * S1;        // S1
  float* YB  = W + 10 * S1;       // S1
  float* Y   = W + 11 * S1;       // S1
  float* DBL = W + 12 * S1;       // BL*24
  float* MEAN = DBL + rows * 24;  // 1024
  float* ATTN = MEAN + BN * 256;  // 1024
  float* FUSED = W + 4 * S1;      // 8*S1, reuses temp region after cross blocks

  // 1) proj + gated 2x2 pool  -> (B, 4096, 128)
  proj_pool_kernel<<<BN * 64 * 64, 128, 0, stream>>>(pan_in, proj_pan_dw,
                                                     proj_pan_pw, gate, PAN);
  proj_pool_kernel<<<BN * 64 * 64, 128, 0, stream>>>(ms_in, proj_ms_dw,
                                                     proj_ms_pw, gate, MS);

  const int rmsBlocks = (int)((rows * 32 + 255) / 256);

  // 2) single_block (shared mamba) on PAN and MS
  {
    float* streams[2] = { PAN, MS };
    const float* norms[2]  = { spa_norm, spe_norm };
    const float* gammas[2] = { spa_gamma, spe_gamma };
    for (int i = 0; i < 2; ++i) {
      float* X = streams[i];
      rmsnorm_kernel<<<rmsBlocks, 256, 0, stream>>>(X, DM, norms[i], XN, DM, (int)rows);
      launch_gemm(stream, XN, DM, p_shared.in_proj, DM, 1, XZ, 2 * DM,
                  (int)rows, 2 * DM, DM);
      run_mamba_core(stream, p_shared, XZ, 2 * DM, XZ + DM, 2 * DM,
                     XC, DT, DBL, YF, YB, Y, XN);
      addscale_kernel<<<grid_e(rows * DM), 256, 0, stream>>>(X, Y, gammas[i], X, rows);
    }
  }

  // 3) cross blocks
  struct CrossArgs {
    const float *n0, *n1; const MambaP* mp;
    const float *fc1, *dww, *dwb, *fc2;
    const float *x0, *x1; float* out;
  } crosses[2] = {
    { spaN0, spaN1, &p_cspa, spa_fc1, spa_dww, spa_dwb, spa_fc2, PAN, MS, SPA },
    { speN0, speN1, &p_cspe, spe_fc1, spe_dww, spe_dwb, spe_fc2, MS, PAN, SPE },
  };
  for (int i = 0; i < 2; ++i) {
    const CrossArgs& ca = crosses[i];
    rmsnorm_kernel<<<rmsBlocks, 256, 0, stream>>>(ca.x0, DM, ca.n0, XN, DM, (int)rows);
    launch_gemm(stream, XN, DM, ca.mp->in_proj, DM, 1, XZ, 2 * DM, (int)rows, DM, DM);
    rmsnorm_kernel<<<rmsBlocks, 256, 0, stream>>>(ca.x1, DM, ca.n1, XN, DM, (int)rows);
    launch_gemm(stream, XN, DM, ca.mp->z_proj, DM, 1, XZ + DM, 2 * DM, (int)rows, DM, DM);
    run_mamba_core(stream, *ca.mp, XZ, 2 * DM, XZ + DM, 2 * DM,
                   XC, DT, DBL, YF, YB, Y, XN);
    addscale_kernel<<<grid_e(rows * DM), 256, 0, stream>>>(ca.x0, Y, nullptr, ca.out, rows);
    // FFN: h = silu(dwconv(x @ fc1^T)) @ fc2^T ; out += h
    launch_gemm(stream, ca.out, DM, ca.fc1, DM, 1, XN, 64, (int)rows, 64, DM);
    dwconv_silu_kernel<<<grid_e(rows * 64), 256, 0, stream>>>(
        XN, 64, 64, ca.dww, 3, 1, ca.dwb, XC, 64, 0);
    launch_gemm(stream, XC, 64, ca.fc2, 64, 1, YF, DM, (int)rows, DM, 64);
    addscale_kernel<<<grid_e(rows * DM), 256, 0, stream>>>(ca.out, YF, nullptr, ca.out, rows);
  }

  // 4) bilinear upsample + concat -> fused (B,256,128,128)
  const size_t fusedN = (size_t)BN * 256 * 128 * 128;
  fuse_resize_kernel<<<grid_e(fusedN), 256, 0, stream>>>(SPA, SPE, FUSED);

  // 5) ECA channel attention
  chanmean_kernel<<<BN * 256, 256, 0, stream>>>(FUSED, MEAN);
  eca_kernel<<<(BN * 256 + 255) / 256, 256, 0, stream>>>(MEAN, eca_w, ATTN);
  scale_fused_kernel<<<grid_e(fusedN), 256, 0, stream>>>(FUSED, ATTN);

  // 6) out_conv: out[b,o,hw] = out_conv[o,:] . fused[b,:,hw]  (WMMA, K=256)
  float* out = (float*)d_out;
  for (int b = 0; b < BN; ++b) {
    launch_gemm(stream, out_conv, 256,
                FUSED + (size_t)b * 256 * 16384, 16384, 0,
                out + (size_t)b * DM * 16384, 16384,
                DM, 16384, 256);
  }
}
